// DecisionTransformer_34248069219166
// MI455X (gfx1250) — matrix operand order repaired and checked by reading, not compiled, not run on hardware
//
#include <hip/hip_runtime.h>

#define B_ 8
#define T_ 512
#define SD_ 17
#define AD_ 6
#define E_ 512
#define H_ 8
#define L_ 3
#define FF_ 2048
#define S3_ 1536
#define D_ 64
#define M_ (B_ * S3_) /* 12288 token rows */

typedef __attribute__((ext_vector_type(16))) _Float16 v16h;
typedef __attribute__((ext_vector_type(8)))  _Float16 v8h;
typedef __attribute__((ext_vector_type(8)))  float    v8f;

// ---------------------------------------------------------------------------
// WMMA helpers (v_wmma_f32_16x16x32_f16), layouts per CDNA5 ISA
// ---------------------------------------------------------------------------
__device__ __forceinline__ v8f wmma16(v16h a, v16h b, v8f c) {
  return __builtin_amdgcn_wmma_f32_16x16x32_f16(false, a, false, b, (short)0, c,
                                                false, false);
}

// A fragment: 16x32 f16 tile, row stride ld. lane<16 -> M=lane, K={0..7,16..23};
// lane>=16 -> M=lane-16, K={8..15,24..31}.
__device__ __forceinline__ v16h load_afrag(const _Float16* base, int ld, int lane) {
  const int row  = lane & 15;
  const int koff = (lane < 16) ? 0 : 8;
  const _Float16* p = base + (size_t)row * ld + koff;
  union { v16h v; v8h h[2]; } u;
  u.h[0] = *(const v8h*)(p);
  u.h[1] = *(const v8h*)(p + 16);
  return u.v;
}

// B fragment: 32x16 tile stored TRANSPOSED in src (src[n*ld + k] = B[k][n]).
// lane col n = lane&15; lane<16 -> K={0..7,16..23}; lane>=16 -> K={8..15,24..31}.
__device__ __forceinline__ v16h load_bfragT(const _Float16* src, int ld, int lane) {
  const int col = lane & 15;
  const int kb2 = (lane < 16) ? 0 : 8;
  const _Float16* p = src + (size_t)col * ld + kb2;
  union { v16h v; v8h h[2]; } u;
  u.h[0] = *(const v8h*)(p);
  u.h[1] = *(const v8h*)(p + 16);
  return u.v;
}

// ---------------------------------------------------------------------------
// Async global -> LDS (CDNA5 ASYNCcnt-tracked DMA). LDS address = low 32 bits
// of the flat shared address (aperture lives in addr[63:32] per ISA 10.2).
// ---------------------------------------------------------------------------
__device__ __forceinline__ void async_ld16(const void* g, void* l) {
  asm volatile("global_load_async_to_lds_b128 %0, %1, off"
               :: "v"((unsigned)(unsigned long long)(uintptr_t)l), "v"(g)
               : "memory");
}

template <int N>
__device__ __forceinline__ void wait_async() {
#if __has_builtin(__builtin_amdgcn_s_wait_asynccnt)
  __builtin_amdgcn_s_wait_asynccnt((unsigned short)N);
#else
  asm volatile("s_wait_asynccnt %0" :: "i"(N) : "memory");
#endif
}

// ---------------------------------------------------------------------------
// One-time weight prep: f32 [nmat][K][N] -> f16 transposed [nmat][N][K]
// ---------------------------------------------------------------------------
__global__ void conv_transpose_kernel(const float* __restrict__ w,
                                      _Float16* __restrict__ wt, int K, int N,
                                      int nmat) {
  const long long idx = (long long)blockIdx.x * blockDim.x + threadIdx.x;
  const long long per = (long long)K * N;
  if (idx >= per * nmat) return;
  const int l = (int)(idx / per);
  const int r = (int)(idx % per);
  const int n = r / K;
  const int k = r % K;
  wt[idx] = (_Float16)w[l * per + (long long)k * N + n];
}

// ---------------------------------------------------------------------------
// Tiled WMMA GEMM: C[M,N] = A[M,K]*W[K,N] + bias ; optional exact-GELU.
// W passed pre-transposed (WT[N][K]). 128 threads = 4 waves, tile 128x64,
// K-step 32, double-buffered async global->LDS staging.
// ---------------------------------------------------------------------------
__global__ __launch_bounds__(128) void gemm_f16_wmma(
    const _Float16* __restrict__ A, const _Float16* __restrict__ WT,
    const float* __restrict__ bias, float* __restrict__ Cf,
    _Float16* __restrict__ Ch, int Ndim, int Kdim, int act) {
  __shared__ _Float16 As[2][128][32];  // 64B rows
  __shared__ _Float16 Bs[2][64][40];   // 80B rows (16B aligned, bank-skewed)
  const int tid  = threadIdx.x;
  const int wave = tid >> 5;
  const int lane = tid & 31;
  const int row0 = blockIdx.y * 128;
  const int col0 = blockIdx.x * 64;

  const _Float16* Arow = A + (size_t)(row0 + tid) * Kdim;  // one row / thread
  const _Float16* Wrow =
      WT + (size_t)(col0 + (tid >> 1)) * Kdim + ((tid & 1) << 4);

  auto stage = [&](int buf, int kk) {
    const _Float16* ag = Arow + kk;
    _Float16* al = &As[buf][tid][0];
    async_ld16(ag,      al);
    async_ld16(ag + 8,  al + 8);
    async_ld16(ag + 16, al + 16);
    async_ld16(ag + 24, al + 24);
    const _Float16* wg = Wrow + kk;
    _Float16* wl = &Bs[buf][tid >> 1][(tid & 1) << 4];
    async_ld16(wg,     wl);
    async_ld16(wg + 8, wl + 8);
  };

  v8f acc[4][2] = { { {}, {} }, { {}, {} }, { {}, {} }, { {}, {} } };

  stage(0, 0);
  int cur = 0;
  for (int kk = 0; kk < Kdim; kk += 32) {
    if (kk + 32 < Kdim) {
      stage(cur ^ 1, kk + 32);  // prefetch next tile while current drains
      wait_async<6>();          // drain older stage only (6 asyncs/wave/stage)
    } else {
      wait_async<0>();
    }
    __syncthreads();
    const v16h af0 = load_afrag(&As[cur][wave * 32][0], 32, lane);
    const v16h af1 = load_afrag(&As[cur][wave * 32 + 16][0], 32, lane);
#pragma unroll
    for (int t = 0; t < 4; ++t) {
      const v16h bf = load_bfragT(&Bs[cur][t * 16][0], 40, lane);
      acc[t][0] = wmma16(af0, bf, acc[t][0]);
      acc[t][1] = wmma16(af1, bf, acc[t][1]);
    }
    __syncthreads();
    cur ^= 1;
  }

  const int colL = lane & 15;
  const int radd = (lane < 16) ? 0 : 8;
#pragma unroll
  for (int t = 0; t < 4; ++t) {
#pragma unroll
    for (int m = 0; m < 2; ++m) {
#pragma unroll
      for (int r = 0; r < 8; ++r) {
        const int gr = row0 + wave * 32 + m * 16 + r + radd;
        const int gc = col0 + t * 16 + colL;
        float v = acc[t][m][r] + bias[gc];
        if (act == 1) v = 0.5f * v * (1.0f + erff(v * 0.70710678118654752f));
        const size_t o = (size_t)gr * Ndim + gc;
        if (Cf) Cf[o] = v;
        if (Ch) Ch[o] = (_Float16)v;
      }
    }
  }
}

// ---------------------------------------------------------------------------
// Flash attention: one wave per (b, head, 16-row Q tile), causal, D=64.
// ---------------------------------------------------------------------------
__global__ __launch_bounds__(32) void attn_flash_wmma(
    const _Float16* __restrict__ Q, const _Float16* __restrict__ K,
    const _Float16* __restrict__ V, _Float16* __restrict__ O) {
  __shared__ _Float16 Ps[16][32];
  __shared__ _Float16 VsT[64][40];
  const int lane = threadIdx.x;
  const int q0 = blockIdx.x * 16;
  const int hh = blockIdx.y;
  const int bb = blockIdx.z;
  const size_t base = (size_t)bb * S3_ * E_ + hh * D_;
  const _Float16* Qb = Q + base + (size_t)q0 * E_;
  const _Float16* Kb = K + base;
  const _Float16* Vb = V + base;

  const v16h aq0 = load_afrag(Qb, E_, lane);       // d 0..31
  const v16h aq1 = load_afrag(Qb + 32, E_, lane);  // d 32..63

  v8f o[4] = { {}, {}, {}, {} };
  float mrun[8], lrun[8];
#pragma unroll
  for (int r = 0; r < 8; ++r) { mrun[r] = -3.0e38f; lrun[r] = 0.0f; }

  const int colL  = lane & 15;
  const int rbase = q0 + ((lane < 16) ? 0 : 8);
  const int nch   = (q0 + 16 + 31) / 32;

  for (int ch = 0; ch < nch; ++ch) {
    const int kb = ch * 32;
    { // stage V chunk (32x64) transposed: VsT[c][r] = V[kb+r][c]
      const v8h* vr = (const v8h*)(Vb + (size_t)(kb + lane) * E_);
#pragma unroll
      for (int cc = 0; cc < 8; ++cc) {
        const v8h chunk = vr[cc];
#pragma unroll
        for (int e = 0; e < 8; ++e) VsT[cc * 8 + e][lane] = chunk[e];
      }
    }
    // scores S(16x32) = Q(16x64) . K^T
    v8f s0 = {}, s1 = {};
    const _Float16* Kc = Kb + (size_t)kb * E_;
    s0 = wmma16(aq0, load_bfragT(Kc, E_, lane), s0);
    s0 = wmma16(aq1, load_bfragT(Kc + 32, E_, lane), s0);
    s1 = wmma16(aq0, load_bfragT(Kc + 16 * E_, E_, lane), s1);
    s1 = wmma16(aq1, load_bfragT(Kc + 16 * E_ + 32, E_, lane), s1);

    float p0[8], p1[8], cm[8], ef[8], ls[8];
    const int c0 = kb + colL, c1 = kb + 16 + colL;
#pragma unroll
    for (int r = 0; r < 8; ++r) {
      const int grow = rbase + r;
      float a  = s0[r] * 0.125f;  // 1/sqrt(64)
      float b2 = s1[r] * 0.125f;
      if (c0 > grow) a  = -1.0e30f;
      if (c1 > grow) b2 = -1.0e30f;
      p0[r] = a; p1[r] = b2;
      cm[r] = fmaxf(a, b2);
    }
#pragma unroll
    for (int m = 1; m < 16; m <<= 1)
#pragma unroll
      for (int r = 0; r < 8; ++r) cm[r] = fmaxf(cm[r], __shfl_xor(cm[r], m, 32));
#pragma unroll
    for (int r = 0; r < 8; ++r) {
      const float mnew = fmaxf(mrun[r], cm[r]);
      ef[r]  = __expf(mrun[r] - mnew);
      mrun[r] = mnew;
      p0[r] = __expf(p0[r] - mnew);
      p1[r] = __expf(p1[r] - mnew);
      ls[r] = p0[r] + p1[r];
    }
#pragma unroll
    for (int m = 1; m < 16; m <<= 1)
#pragma unroll
      for (int r = 0; r < 8; ++r) ls[r] += __shfl_xor(ls[r], m, 32);
#pragma unroll
    for (int r = 0; r < 8; ++r) {
      lrun[r] = lrun[r] * ef[r] + ls[r];
#pragma unroll
      for (int t = 0; t < 4; ++t) o[t][r] = o[t][r] * ef[r];
    }
    { // write P tile (C-layout -> LDS row-major 16x32)
      const int rowoff = (lane < 16) ? 0 : 8;
#pragma unroll
      for (int r = 0; r < 8; ++r) {
        Ps[r + rowoff][colL]      = (_Float16)p0[r];
        Ps[r + rowoff][16 + colL] = (_Float16)p1[r];
      }
    }
    __syncthreads();
    const v16h ap = load_afrag(&Ps[0][0], 32, lane);
#pragma unroll
    for (int t = 0; t < 4; ++t)
      o[t] = wmma16(ap, load_bfragT(&VsT[t * 16][0], 40, lane), o[t]);
    __syncthreads();
  }

  const int radd = (lane < 16) ? 0 : 8;
#pragma unroll
  for (int r = 0; r < 8; ++r) {
    const float inv = 1.0f / lrun[r];
#pragma unroll
    for (int t = 0; t < 4; ++t)
      O[base + (size_t)(q0 + r + radd) * E_ + t * 16 + colL] =
          (_Float16)(o[t][r] * inv);
  }
}

// ---------------------------------------------------------------------------
// Fused embedding (reward/state/action proj + time emb) + LayerNorm
// ---------------------------------------------------------------------------
__global__ __launch_bounds__(512) void embed_ln_kernel(
    const float* __restrict__ state, const float* __restrict__ action,
    const float* __restrict__ reward, const int* __restrict__ timestep,
    const float* __restrict__ sW, const float* __restrict__ sb,
    const float* __restrict__ aW, const float* __restrict__ ab,
    const float* __restrict__ rW, const float* __restrict__ rb,
    const float* __restrict__ temb, const float* __restrict__ g,
    const float* __restrict__ be, float* __restrict__ hf,
    _Float16* __restrict__ hh) {
  __shared__ float red[512];
  const int row = blockIdx.x;
  const int e = threadIdx.x;
  const int b = row / S3_;
  const int s3 = row % S3_;
  const int t = s3 / 3;
  const int j = s3 % 3;
  const int bt = b * T_ + t;
  float x;
  if (j == 0) {
    x = reward[bt] * rW[e] + rb[e];
  } else if (j == 1) {
    x = sb[e];
    const float* sp = state + (size_t)bt * SD_;
#pragma unroll
    for (int i = 0; i < SD_; ++i) x += sp[i] * sW[i * E_ + e];
  } else {
    x = ab[e];
    const float* ap = action + (size_t)bt * AD_;
#pragma unroll
    for (int i = 0; i < AD_; ++i) x += ap[i] * aW[i * E_ + e];
  }
  x += temb[(size_t)timestep[bt] * E_ + e];

  red[e] = x; __syncthreads();
  for (int s = 256; s > 0; s >>= 1) { if (e < s) red[e] += red[e + s]; __syncthreads(); }
  const float mean = red[0] * (1.0f / E_);
  __syncthreads();
  const float d = x - mean;
  red[e] = d * d; __syncthreads();
  for (int s = 256; s > 0; s >>= 1) { if (e < s) red[e] += red[e + s]; __syncthreads(); }
  const float var = red[0] * (1.0f / E_);
  const float y = d * rsqrtf(var + 1e-5f) * g[e] + be[e];
  hf[(size_t)row * E_ + e] = y;
  hh[(size_t)row * E_ + e] = (_Float16)y;
}

// ---------------------------------------------------------------------------
// Residual add + LayerNorm (in-place safe on hf)
// ---------------------------------------------------------------------------
__global__ __launch_bounds__(512) void residual_ln_kernel(
    const float* __restrict__ basein, const float* __restrict__ delta,
    const float* __restrict__ g, const float* __restrict__ be,
    float* __restrict__ hf, _Float16* __restrict__ hh) {
  __shared__ float red[512];
  const int row = blockIdx.x;
  const int e = threadIdx.x;
  const size_t idx = (size_t)row * E_ + e;
  const float x = basein[idx] + delta[idx];
  red[e] = x; __syncthreads();
  for (int s = 256; s > 0; s >>= 1) { if (e < s) red[e] += red[e + s]; __syncthreads(); }
  const float mean = red[0] * (1.0f / E_);
  __syncthreads();
  const float d = x - mean;
  red[e] = d * d; __syncthreads();
  for (int s = 256; s > 0; s >>= 1) { if (e < s) red[e] += red[e + s]; __syncthreads(); }
  const float var = red[0] * (1.0f / E_);
  const float y = d * rsqrtf(var + 1e-5f) * g[e] + be[e];
  hf[idx] = y;
  hh[idx] = (_Float16)y;
}

// ---------------------------------------------------------------------------
// Output heads: state_preds | action_preds | return_preds (concatenated)
// ---------------------------------------------------------------------------
__global__ void heads_kernel(const float* __restrict__ hf,
                             const float* __restrict__ RW, const float* __restrict__ Rb,
                             const float* __restrict__ SW, const float* __restrict__ Sb,
                             const float* __restrict__ AW, const float* __restrict__ Ab,
                             float* __restrict__ out) {
  const int NSP = B_ * T_ * SD_;
  const int NAP = B_ * T_ * AD_;
  const int NRP = B_ * T_;
  const int idx = blockIdx.x * blockDim.x + threadIdx.x;
  if (idx >= NSP + NAP + NRP) return;
  if (idx < NSP) {
    const int s = idx % SD_;
    const int bt = idx / SD_;
    const int b = bt / T_, t = bt % T_;
    const float* hrow = hf + ((size_t)b * S3_ + 3 * t + 2) * E_;  // h_action
    float acc = Sb[s];
    for (int e = 0; e < E_; ++e) acc += hrow[e] * SW[e * SD_ + s];
    out[idx] = acc;
  } else if (idx < NSP + NAP) {
    const int k = idx - NSP;
    const int a = k % AD_;
    const int bt = k / AD_;
    const int b = bt / T_, t = bt % T_;
    const float* hrow = hf + ((size_t)b * S3_ + 3 * t + 1) * E_;  // h_state
    float acc = Ab[a];
    for (int e = 0; e < E_; ++e) acc += hrow[e] * AW[e * AD_ + a];
    out[idx] = tanhf(acc);
  } else {
    const int k = idx - NSP - NAP;
    const int b = k / T_, t = k % T_;
    const float* hrow = hf + ((size_t)b * S3_ + 3 * t + 2) * E_;  // h_action
    float acc = Rb[0];
    for (int e = 0; e < E_; ++e) acc += hrow[e] * RW[e];
    out[idx] = acc;
  }
}

// ---------------------------------------------------------------------------
// Host orchestration
// ---------------------------------------------------------------------------
extern "C" void kernel_launch(void* const* d_in, const int* in_sizes, int n_in,
                              void* d_out, int out_size, void* d_ws, size_t ws_size,
                              hipStream_t stream) {
  (void)in_sizes; (void)n_in; (void)out_size; (void)ws_size;
  const float* state   = (const float*)d_in[0];
  const float* action  = (const float*)d_in[1];
  const float* reward  = (const float*)d_in[2];
  const int*   tstep   = (const int*)d_in[3];
  const float* sW  = (const float*)d_in[4];
  const float* sb  = (const float*)d_in[5];
  const float* aW  = (const float*)d_in[6];
  const float* ab  = (const float*)d_in[7];
  const float* rW  = (const float*)d_in[8];
  const float* rb  = (const float*)d_in[9];
  const float* temb = (const float*)d_in[10];
  const float* lnEg = (const float*)d_in[11];
  const float* lnEb = (const float*)d_in[12];
  const float* qW  = (const float*)d_in[13];
  const float* qb  = (const float*)d_in[14];
  const float* kW  = (const float*)d_in[15];
  const float* kb  = (const float*)d_in[16];
  const float* vW  = (const float*)d_in[17];
  const float* vb  = (const float*)d_in[18];
  const float* pW  = (const float*)d_in[19];
  const float* pb  = (const float*)d_in[20];
  const float* ln1g = (const float*)d_in[21];
  const float* ln1b = (const float*)d_in[22];
  const float* m1W = (const float*)d_in[23];
  const float* m1b = (const float*)d_in[24];
  const float* m2W = (const float*)d_in[25];
  const float* m2b = (const float*)d_in[26];
  const float* ln2g = (const float*)d_in[27];
  const float* ln2b = (const float*)d_in[28];
  const float* hdRW = (const float*)d_in[29];
  const float* hdRb = (const float*)d_in[30];
  const float* hdSW = (const float*)d_in[31];
  const float* hdSb = (const float*)d_in[32];
  const float* hdAW = (const float*)d_in[33];
  const float* hdAb = (const float*)d_in[34];
  float* out = (float*)d_out;

  // workspace partitioning (256B aligned)
  char* wsp = (char*)d_ws;
  auto alloc = [&](size_t bytes) -> void* {
    void* r = (void*)wsp;
    wsp += (bytes + 255) & ~(size_t)255;
    return r;
  };
  float*    hf   = (float*)   alloc((size_t)M_ * E_ * 4);
  _Float16* hh   = (_Float16*)alloc((size_t)M_ * E_ * 2);
  _Float16* qh   = (_Float16*)alloc((size_t)M_ * E_ * 2);
  _Float16* kh   = (_Float16*)alloc((size_t)M_ * E_ * 2);
  _Float16* vh   = (_Float16*)alloc((size_t)M_ * E_ * 2);
  _Float16* atth = (_Float16*)alloc((size_t)M_ * E_ * 2);
  float*    tmp  = (float*)   alloc((size_t)M_ * E_ * 4);
  _Float16* hid  = (_Float16*)alloc((size_t)M_ * FF_ * 2);
  _Float16* wq   = (_Float16*)alloc((size_t)L_ * E_ * E_ * 2);
  _Float16* wk   = (_Float16*)alloc((size_t)L_ * E_ * E_ * 2);
  _Float16* wv   = (_Float16*)alloc((size_t)L_ * E_ * E_ * 2);
  _Float16* wp   = (_Float16*)alloc((size_t)L_ * E_ * E_ * 2);
  _Float16* w1   = (_Float16*)alloc((size_t)L_ * E_ * FF_ * 2);
  _Float16* w2   = (_Float16*)alloc((size_t)L_ * FF_ * E_ * 2);

  // one-time transposed f16 weight prep (WT[N][K] per layer)
  auto convT = [&](const float* s, _Float16* d, int K, int N) {
    const long long tot = (long long)L_ * K * N;
    conv_transpose_kernel<<<(int)((tot + 255) / 256), 256, 0, stream>>>(s, d, K,
                                                                        N, L_);
  };
  convT(qW,  wq, E_, E_);
  convT(kW,  wk, E_, E_);
  convT(vW,  wv, E_, E_);
  convT(pW,  wp, E_, E_);
  convT(m1W, w1, E_, FF_);
  convT(m2W, w2, FF_, E_);

  embed_ln_kernel<<<M_, 512, 0, stream>>>(state, action, reward, tstep, sW, sb,
                                          aW, ab, rW, rb, temb, lnEg, lnEb, hf, hh);

  auto gemm = [&](const _Float16* A, const _Float16* WT, const float* bias,
                  float* Cf, _Float16* Ch, int Ndim, int Kdim, int act) {
    dim3 g(Ndim / 64, M_ / 128);
    gemm_f16_wmma<<<g, 128, 0, stream>>>(A, WT, bias, Cf, Ch, Ndim, Kdim, act);
  };

  for (int l = 0; l < L_; ++l) {
    const size_t wo = (size_t)l * E_ * E_;
    gemm(hh, wq + wo, qb + l * E_, nullptr, qh, E_, E_, 0);
    gemm(hh, wk + wo, kb + l * E_, nullptr, kh, E_, E_, 0);
    gemm(hh, wv + wo, vb + l * E_, nullptr, vh, E_, E_, 0);

    attn_flash_wmma<<<dim3(S3_ / 16, H_, B_), 32, 0, stream>>>(qh, kh, vh, atth);

    gemm(atth, wp + wo, pb + l * E_, tmp, nullptr, E_, E_, 0);
    residual_ln_kernel<<<M_, 512, 0, stream>>>(hf, tmp, ln1g + l * E_,
                                               ln1b + l * E_, hf, hh);

    gemm(hh, w1 + (size_t)l * E_ * FF_, m1b + l * FF_, nullptr, hid, FF_, E_, 1);
    gemm(hid, w2 + (size_t)l * FF_ * E_, m2b + l * E_, tmp, nullptr, E_, FF_, 0);
    residual_ln_kernel<<<M_, 512, 0, stream>>>(hf, tmp, ln2g + l * E_,
                                               ln2b + l * E_, hf, hh);
  }

  const int ntot = B_ * T_ * (SD_ + AD_ + 1);
  heads_kernel<<<(ntot + 255) / 256, 256, 0, stream>>>(hf, hdRW, hdRb, hdSW,
                                                       hdSb, hdAW, hdAb, out);
}
